// PILayer_86122684219992
// MI455X (gfx1250) — matrix operand order repaired and checked by reading, not compiled, not run on hardware
//
#include <hip/hip_runtime.h>

// CDNA5 / gfx1250 WMMA types
typedef __attribute__((ext_vector_type(16))) __bf16 v16bf;
typedef __attribute__((ext_vector_type(8)))  float  v8f;

#define F_DIM       64
#define B_DIM       16
#define KDIM        128            // 2*F_DIM (concat of two node rows)
#define NDIM        1024           // F_DIM*B_DIM (MLP out)
#define NT_COUNT    64             // NDIM / 16 tiles == one per output feature
#define KSTEPS      4              // KDIM / 32 (bf16 WMMA K = 32)
#define BLOCK_EDGES 64             // 4 edge-subtiles of 16
#define THREADS     256            // 8 wave32

// Native hardware tanh on gfx1250 (V_TANH_F32, single TRANS32 op).
__device__ __forceinline__ float fast_tanh(float x) {
#if defined(__HIP_DEVICE_COMPILE__) && __has_builtin(__builtin_amdgcn_tanhf)
    return __builtin_amdgcn_tanhf(x);
#else
    x = fminf(fmaxf(x, -15.0f), 15.0f);
    float e = __expf(2.0f * x);
    return (e - 1.0f) / (e + 1.0f);
#endif
}

// ---------------------------------------------------------------------------
// TRANSPOSED GEMM: D = A x B with A = W-tile^T (M = MLP output column,
// K = input feature), B = inter^T (K x 16 edges). Then D[b][e] = H[e][b]:
//   lane -> edge (col N), VGPR r -> b = r + (lane>>4)*8 (row M)
// so the basis contraction over b is a per-lane register reduction plus one
// cross-half shuffle — no serialized ds_bpermute chains.
//
// Fragment layouts used here (contiguous k order, not the ISA table order; wave32):
//  A (16x32 bf16): lane l -> row M = l&15; g = l>>4:
//      element e -> k = g*16 + e
//  B (32x16 bf16): lane l -> col N = l&15; g = l>>4: element e -> k = g*16 + e
// ---------------------------------------------------------------------------

// Prep: pack W [128,1024] row-major f32 into bf16 WMMA *A* fragments (W^T
// tiles) in d_ws. Fragment (nt, ks): 32 lanes x 16 bf16 = 1 KB; 256 KB total.
__global__ void pack_W_bf16(const float* __restrict__ W, v16bf* __restrict__ Wp) {
    int t = blockIdx.x * blockDim.x + threadIdx.x;      // 0 .. 8191 lane-slots
    if (t >= NT_COUNT * KSTEPS * 32) return;
    int lane = t & 31;
    int ks   = (t >> 5) & 3;
    int nt   = t >> 7;
    int g    = lane >> 4;
    int n    = nt * 16 + (lane & 15);                   // A row m == W column n
    v16bf frag;
#pragma unroll
    for (int e = 0; e < 16; ++e) {
        int kl = g * 16 + e;
        frag[e] = (__bf16)W[(ks * 32 + kl) * NDIM + n];
    }
    Wp[t] = frag;
}

__global__ __launch_bounds__(THREADS)
void pilayer_wmma_kernel(const float* __restrict__ p1,
                         const int*   __restrict__ pair_i,
                         const int*   __restrict__ pair_j,
                         const float* __restrict__ basis,
                         const float* __restrict__ bias,
                         const v16bf* __restrict__ Wp,
                         float*       __restrict__ out,
                         int E) {
    // LDS: B fragments (inter^T, 4 msub x 4 ks x 32 lanes x 16 bf16 = 16 KB)
    //      + output tile (64x64 f32 = 16 KB).
    __shared__ __align__(32) __bf16 Blds[4 * KSTEPS * 32 * 16];
    __shared__ float out_tile[BLOCK_EDGES * F_DIM];

    const int  tid        = threadIdx.x;
    const long blockStart = (long)blockIdx.x * BLOCK_EDGES;

    // ---- stage gathered inter^T as WMMA B fragments ----
    // lane-slot s = (msub, ks, lane): lane -> edge col; each lane reads 16
    // CONSECUTIVE floats of its edge's row (ks 0..1: p1[pair_i], 2..3: p1[pair_j]).
    for (int s = tid; s < 4 * KSTEPS * 32; s += THREADS) {
        int lane = s & 31;
        int ks   = (s >> 5) & 3;
        int msub = s >> 7;
        int g    = lane >> 4;
        long edge = blockStart + msub * 16 + (lane & 15);
        if (edge >= E) edge = E - 1;                    // clamp for tail block
        int src = (ks < 2) ? pair_i[edge] : pair_j[edge];
        const float* row = p1 + (long)src * F_DIM + (ks & 1) * 32 + g * 16;
        float4 a0 = *(const float4*)(row + 0);
        float4 a1 = *(const float4*)(row + 4);
        float4 a2 = *(const float4*)(row + 8);
        float4 a3 = *(const float4*)(row + 12);
        __bf16* dst = &Blds[s * 16];
        dst[0]  = (__bf16)a0.x; dst[1]  = (__bf16)a0.y;
        dst[2]  = (__bf16)a0.z; dst[3]  = (__bf16)a0.w;
        dst[4]  = (__bf16)a1.x; dst[5]  = (__bf16)a1.y;
        dst[6]  = (__bf16)a1.z; dst[7]  = (__bf16)a1.w;
        dst[8]  = (__bf16)a2.x; dst[9]  = (__bf16)a2.y;
        dst[10] = (__bf16)a2.z; dst[11] = (__bf16)a2.w;
        dst[12] = (__bf16)a3.x; dst[13] = (__bf16)a3.y;
        dst[14] = (__bf16)a3.z; dst[15] = (__bf16)a3.w;
    }

    const int wave = tid >> 5;
    const int lane = tid & 31;
    const int g    = lane >> 4;

    // ---- preload per-lane basis weights: 8 contiguous floats per msub ----
    // lane's edge e = lane&15; r-th accumulator row is b = g*8 + r.
    float basw[4][8];
#pragma unroll
    for (int msub = 0; msub < 4; ++msub) {
        long edge = blockStart + msub * 16 + (lane & 15);
        if (edge >= E) edge = E - 1;
        const float* bp = basis + edge * B_DIM + g * 8;
        float4 lo = *(const float4*)(bp + 0);
        float4 hi = *(const float4*)(bp + 4);
        basw[msub][0] = lo.x; basw[msub][1] = lo.y;
        basw[msub][2] = lo.z; basw[msub][3] = lo.w;
        basw[msub][4] = hi.x; basw[msub][5] = hi.y;
        basw[msub][6] = hi.z; basw[msub][7] = hi.w;
    }
    __syncthreads();

    // Each wave owns 8 tiles nt (output features); W fragments loaded once
    // per tile, reused over the 4 edge-subtiles (4 WMMAs each, K = 128).
    for (int ni = 0; ni < 8; ++ni) {
        const int nt = wave + ni * 8;
        v16bf Af[KSTEPS];
#pragma unroll
        for (int ks = 0; ks < KSTEPS; ++ks)
            Af[ks] = Wp[(nt * KSTEPS + ks) * 32 + lane];

        // bias for rows b = g*8 .. g*8+7 of this tile (broadcast loads)
        const float* bb = bias + nt * 16 + g * 8;
        float4 bl = *(const float4*)(bb + 0);
        float4 bh = *(const float4*)(bb + 4);
        const float biasr[8] = {bl.x, bl.y, bl.z, bl.w, bh.x, bh.y, bh.z, bh.w};

        for (int msub = 0; msub < 4; ++msub) {
            v16bf Bf[KSTEPS];
#pragma unroll
            for (int ks = 0; ks < KSTEPS; ++ks)
                Bf[ks] = *(const v16bf*)&Blds[((msub * KSTEPS + ks) * 32 + lane) * 16];

            v8f acc = {};
#pragma unroll
            for (int ks = 0; ks < KSTEPS; ++ks)
                acc = __builtin_amdgcn_wmma_f32_16x16x32_bf16(
                          false, Af[ks], false, Bf[ks],
                          (short)0, acc, false, false);

            // Fused epilogue: bias + hw tanh + basis weight, reduced over b
            // entirely in registers; one cross-half shuffle combines halves.
            float s = 0.0f;
#pragma unroll
            for (int r = 0; r < 8; ++r)
                s = fmaf(fast_tanh(acc[r] + biasr[r]), basw[msub][r], s);
            s += __shfl_xor(s, 16, 32);                 // b-halves: g=0 + g=1
            if (lane < 16)
                out_tile[(msub * 16 + lane) * F_DIM + nt] = s;
        }
    }
    __syncthreads();

    // ---- coalesced writeback of the 64x64 f32 tile ----
    for (int q = tid; q < (BLOCK_EDGES * F_DIM) / 4; q += THREADS) {
        long edge = blockStart + (q >> 4);              // 16 float4 per edge row
        if (edge < E)
            ((float4*)out)[edge * (F_DIM / 4) + (q & 15)] =
                ((const float4*)out_tile)[q];
    }
}

extern "C" void kernel_launch(void* const* d_in, const int* in_sizes, int n_in,
                              void* d_out, int out_size, void* d_ws, size_t ws_size,
                              hipStream_t stream) {
    const float* p1     = (const float*)d_in[0];
    const int*   pair_i = (const int*)  d_in[1];
    const int*   pair_j = (const int*)  d_in[2];
    const float* basis  = (const float*)d_in[3];
    const float* W      = (const float*)d_in[4];
    const float* bias   = (const float*)d_in[5];
    float*       out    = (float*)d_out;
    const int E = in_sizes[1];

    v16bf* Wp = (v16bf*)d_ws;                           // 256 KB, L2-resident

    // 1) repack W^T to bf16 WMMA A-fragments (8192 lane-slots)
    pack_W_bf16<<<(NT_COUNT * KSTEPS * 32 + 255) / 256, 256, 0, stream>>>(W, Wp);

    // 2) fused gather + transposed GEMM(WMMA bf16) + bias + hw-tanh +
    //    register-space basis contraction
    int nBlocks = (E + BLOCK_EDGES - 1) / BLOCK_EDGES;
    pilayer_wmma_kernel<<<nBlocks, THREADS, 0, stream>>>(
        p1, pair_i, pair_j, basis, bias, Wp, out, E);
}
